// AttentiveFP_DGL_80418967650695
// MI455X (gfx1250) — matrix operand order repaired and checked by reading, not compiled, not run on hardware
//
#include <hip/hip_runtime.h>
#include <hip/hip_bf16.h>

// ---------------------------------------------------------------------------
// AttentiveFP forward for MI455X (gfx1250, wave32).
// Dense GEMMs -> v_wmma_f32_16x16x32_bf16 (bf16 operands, f32 accumulate),
// with LDS tiles stored directly in WMMA fragment order so each lane reads
// its 32B fragment with wide ds loads. Streaming edge tensors use NT cache
// hints (192MB L2: node tensors fit, 410MB edge tensors do not).
// ---------------------------------------------------------------------------

#define N_NODES  50000
#define N_EDGES  400000
#define N_GRAPHS 1000
#define D_NODE   64
#define D_EDGE   16
#define D_G      256
#define NUM_T    2

typedef __attribute__((ext_vector_type(16))) __bf16 v16bf;
typedef __attribute__((ext_vector_type(8)))  float  v8f;

#if __has_builtin(__builtin_amdgcn_cvt_pk_bf16_f32)
__device__ __forceinline__ unsigned pack_bf16x2(float x, float y) {
    return __builtin_bit_cast(unsigned, __builtin_amdgcn_cvt_pk_bf16_f32(x, y));
}
#else
__device__ __forceinline__ unsigned short f32_to_bf16_bits(float f) {
    unsigned u = __builtin_bit_cast(unsigned, f);
    unsigned r = u + 0x7FFFu + ((u >> 16) & 1u);   // round-to-nearest-even
    return (unsigned short)(r >> 16);
}
__device__ __forceinline__ unsigned pack_bf16x2(float x, float y) {
    return ((unsigned)f32_to_bf16_bits(y) << 16) | (unsigned)f32_to_bf16_bits(x);
}
#endif

// monotonic encoding of float -> uint for atomicMax-based segment max
__device__ __forceinline__ unsigned enc_f(float f) {
    unsigned u = __builtin_bit_cast(unsigned, f);
    return (u & 0x80000000u) ? ~u : (u | 0x80000000u);
}
__device__ __forceinline__ float dec_f(unsigned e) {
    unsigned u = (e & 0x80000000u) ? (e & 0x7FFFFFFFu) : ~e;
    return __builtin_bit_cast(float, u);
}

// ---------------------------------------------------------------------------
// WMMA GEMM: C[M,N] = act(A[M,K] @ B + bias)
//   a_idx != nullptr : row m of A is A[a_idx[m]]     (gather GEMM)
//   b_trans == 1     : B stored [N,K] row-major (x @ W.T for GRU weights)
//   act: 0 = none, 1 = leaky_relu(0.01), 2 = relu
//   nt_c: non-temporal C stores (write-once streamed outputs)
// Block = 128 threads = 4 waves. Block tile 16(M) x 128(N); each wave owns a
// 16x32 C tile (2 accumulators, 2 WMMAs per K-step). K stepped by 32.
// LDS holds tiles in WMMA *fragment* order:
//   A elem (m,k): lane = m + 16*(k>>4), half = k&15
//   B elem (k,n): subtile = n>>4, lane = (n&15) + 16*(k>>4), half = k&15
// K-aligned quads never cross a lane boundary -> one b64 LDS store per quad;
// each lane then reads its whole fragment as an aligned 32B vector.
// Requires N % 128 == 0 (true: 256, 768) and K % 4 == 0 (64, 80, 256).
// ---------------------------------------------------------------------------
__global__ void __launch_bounds__(128)
gemm_wmma_bf16(const float* __restrict__ A, const int* __restrict__ a_idx,
               const float* __restrict__ B, int b_trans,
               const float* __restrict__ bias, float* __restrict__ C,
               int M, int N, int K, int act, int nt_c)
{
    __shared__ alignas(32) unsigned short AsF[32][16];     // A fragments (1 KB)
    __shared__ alignas(32) unsigned short BsF[8][32][16];  // B fragments (8 KB)

    const int tid  = threadIdx.x;
    const int wave = tid >> 5;
    const int lane = tid & 31;
    const int m0   = blockIdx.y * 16;
    const int nblk = blockIdx.x * 128;

    // ---- A loader assignment (hoisted out of the K loop) ----
    const int  am   = tid >> 3;          // row 0..15
    const int  akb  = (tid & 7) * 4;     // k-quad base 0..28
    const int  agm  = m0 + am;
    const bool a_ok = agm < M;
    long arow = 0;
    if (a_ok) arow = a_idx ? (long)a_idx[agm] : (long)agm;
    const float* __restrict__ pA = A + arow * (long)K;
    const int a_lane = am + 16 * (akb >> 4);
    const int a_half = akb & 15;

    // ---- B loader assignment: thread owns column n = nblk + tid, all 32 k ----
    const int bn   = nblk + tid;
    const int bs   = tid >> 4;           // n-subtile 0..7
    const int bn15 = tid & 15;

    v8f acc[2];
    acc[0] = (v8f){0.f,0.f,0.f,0.f,0.f,0.f,0.f,0.f};
    acc[1] = acc[0];

    for (int k0 = 0; k0 < K; k0 += 32) {
        // ---- A tile -> fragment-order LDS (1 float4 load, 1 b64 store) ----
        if (a_ok) __builtin_prefetch(pA + k0 + 32 + akb, 0, 3);  // speculative
        float4 a4 = make_float4(0.f, 0.f, 0.f, 0.f);
        if (a_ok && (k0 + akb) < K)
            a4 = *(const float4*)(pA + k0 + akb);
        uint2 apk;
        apk.x = pack_bf16x2(a4.x, a4.y);
        apk.y = pack_bf16x2(a4.z, a4.w);
        *(uint2*)&AsF[a_lane][a_half] = apk;

        // ---- B tile -> fragment-order LDS (8 k-quads per thread) ----
        if (b_trans) {
            const float* __restrict__ pB = B + (long)bn * K + k0;
            #pragma unroll
            for (int q = 0; q < 8; ++q) {
                const int kq = q * 4;
                float4 b4 = make_float4(0.f, 0.f, 0.f, 0.f);
                if (k0 + kq < K) b4 = *(const float4*)(pB + kq);
                uint2 bpk;
                bpk.x = pack_bf16x2(b4.x, b4.y);
                bpk.y = pack_bf16x2(b4.z, b4.w);
                *(uint2*)&BsF[bs][bn15 + 16 * (kq >> 4)][kq & 15] = bpk;
            }
        } else {
            #pragma unroll
            for (int q = 0; q < 8; ++q) {
                const int kq = q * 4;
                float4 b4 = make_float4(0.f, 0.f, 0.f, 0.f);
                if (k0 + kq < K) {
                    const float* __restrict__ pB = B + (long)(k0 + kq) * N + bn;
                    b4.x = pB[0];
                    b4.y = pB[N];
                    b4.z = pB[2 * (long)N];
                    b4.w = pB[3 * (long)N];
                }
                uint2 bpk;
                bpk.x = pack_bf16x2(b4.x, b4.y);
                bpk.y = pack_bf16x2(b4.z, b4.w);
                *(uint2*)&BsF[bs][bn15 + 16 * (kq >> 4)][kq & 15] = bpk;
            }
        }
        __syncthreads();

        // ---- fragment reads: aligned 32B vector loads, then 2 WMMAs ----
        v16bf av  = ((const v16bf*)AsF)[lane];
        v16bf bv0 = ((const v16bf*)BsF)[(2 * wave)     * 32 + lane];
        v16bf bv1 = ((const v16bf*)BsF)[(2 * wave + 1) * 32 + lane];
        acc[0] = __builtin_amdgcn_wmma_f32_16x16x32_bf16(
                     false, av, false, bv0, (short)0, acc[0], false, false);
        acc[1] = __builtin_amdgcn_wmma_f32_16x16x32_bf16(
                     false, av, false, bv1, (short)0, acc[1], false, false);
        __syncthreads();
    }

    // ---- epilogue: bias + activation; C layout VGPR r -> M = r + 8*(lane>>4)
    const int mloc = lane & 15;
    const int hi   = lane >> 4;
    #pragma unroll
    for (int s = 0; s < 2; ++s) {
        const int n = nblk + (2 * wave + s) * 16 + mloc;
        const float bval = bias ? bias[n] : 0.f;
        #pragma unroll
        for (int r = 0; r < 8; ++r) {
            const int m = m0 + r + 8 * hi;
            if (m < M) {
                float v = acc[s][r] + bval;
                if (act == 1)      v = v > 0.f ? v : 0.01f * v;
                else if (act == 2) v = fmaxf(v, 0.f);
                if (nt_c) __builtin_nontemporal_store(v, &C[(long)m * N + n]);
                else      C[(long)m * N + n] = v;
            }
        }
    }
}

// ---------------------------------------------------------------------------
// Wave-per-row logit: out[r] = lrelu( dot(concat(PA[idxA[r]], PB[idxB[r]]), W) + b )
// 512-dim dot (two 256-wide halves), float4-vectorized, shfl_xor reduce.
// ---------------------------------------------------------------------------
__global__ void __launch_bounds__(256)
row_logit(const float* __restrict__ PA, const int* __restrict__ idxA, int reluA,
          const float* __restrict__ PB, const int* __restrict__ idxB,
          const float* __restrict__ W, const float* __restrict__ b,
          float* __restrict__ out, int rows)
{
    int row = blockIdx.x * 8 + (threadIdx.x >> 5);
    if (row >= rows) return;
    int lane = threadIdx.x & 31;
    int ra = idxA ? idxA[row] : row;
    int rb = idxB ? idxB[row] : row;
    const float4* pa = (const float4*)(PA + (long)ra * D_G);
    const float4* pb = (const float4*)(PB + (long)rb * D_G);
    const float4* wa = (const float4*)W;
    const float4* wb = (const float4*)(W + D_G);
    float s = 0.f;
    #pragma unroll
    for (int i = lane; i < D_G / 4; i += 32) {
        float4 va = pa[i];
        if (reluA) {
            va.x = fmaxf(va.x, 0.f); va.y = fmaxf(va.y, 0.f);
            va.z = fmaxf(va.z, 0.f); va.w = fmaxf(va.w, 0.f);
        }
        float4 w0 = wa[i];
        s = fmaf(va.x, w0.x, s); s = fmaf(va.y, w0.y, s);
        s = fmaf(va.z, w0.z, s); s = fmaf(va.w, w0.w, s);
        float4 vb = pb[i];
        float4 w1 = wb[i];
        s = fmaf(vb.x, w1.x, s); s = fmaf(vb.y, w1.y, s);
        s = fmaf(vb.z, w1.z, s); s = fmaf(vb.w, w1.w, s);
    }
    #pragma unroll
    for (int off = 16; off > 0; off >>= 1) s += __shfl_xor(s, off, 32);
    if (lane == 0) {
        float v = s + b[0];
        out[row] = v > 0.f ? v : 0.01f * v;
    }
}

// ---------------------------------------------------------------------------
// fills / segment softmax / scatter / elementwise
// ---------------------------------------------------------------------------
__global__ void fill_f32(float* p, float v, long n) {
    long i = (long)blockIdx.x * blockDim.x + threadIdx.x;
    if (i < n) p[i] = v;
}
__global__ void fill_u32(unsigned* p, unsigned v, long n) {
    long i = (long)blockIdx.x * blockDim.x + threadIdx.x;
    if (i < n) p[i] = v;
}
__global__ void seg_max_k(const float* __restrict__ logit, const int* __restrict__ seg,
                          unsigned* __restrict__ mx, int n) {
    int i = blockIdx.x * blockDim.x + threadIdx.x;
    if (i < n) atomicMax(&mx[seg[i]], enc_f(logit[i]));
}
__global__ void seg_expsum_k(const float* __restrict__ logit, const int* __restrict__ seg,
                             const unsigned* __restrict__ mx, float* __restrict__ sum,
                             float* __restrict__ ev, int n) {
    int i = blockIdx.x * blockDim.x + threadIdx.x;
    if (i < n) {
        float e = __expf(logit[i] - dec_f(mx[seg[i]]));
        ev[i] = e;
        atomicAdd(&sum[seg[i]], e);
    }
}
__global__ void seg_norm_k(const float* __restrict__ ev, const int* __restrict__ seg,
                           const float* __restrict__ sum, float* __restrict__ a, int n) {
    int i = blockIdx.x * blockDim.x + threadIdx.x;
    if (i < n) a[i] = ev[i] / sum[seg[i]];
}
// out[seg[row]*C + c] += (scale?scale[row]:1) * src[row*C + c]
// nt_src: streaming (read-once) source -> non-temporal loads
__global__ void scatter_scale_add(float* __restrict__ out, const int* __restrict__ seg,
                                  const float* __restrict__ scale,
                                  const float* __restrict__ src, long rows, int C,
                                  int nt_src) {
    long e = (long)blockIdx.x * blockDim.x + threadIdx.x;
    if (e >= rows * (long)C) return;
    long row = e / C;
    int  c   = (int)(e - row * C);
    float v = nt_src ? __builtin_nontemporal_load(&src[e]) : src[e];
    if (scale) v *= scale[row];
    atomicAdd(&out[(long)seg[row] * C + c], v);
}
__global__ void elu_inplace(float* p, long n) {
    long i = (long)blockIdx.x * blockDim.x + threadIdx.x;
    if (i < n) { float x = p[i]; p[i] = x > 0.f ? x : (__expf(x) - 1.f); }
}
// concat input for pe1: X[e,0:64]=hv[src[e]], X[e,64:80]=ef[e]
__global__ void concat_src_edge(const float* __restrict__ hv, const float* __restrict__ ef,
                                const int* __restrict__ src, float* __restrict__ X) {
    long e = (long)blockIdx.x * blockDim.x + threadIdx.x;
    const long Kc = D_NODE + D_EDGE;
    if (e >= (long)N_EDGES * Kc) return;
    long row = e / Kc;
    int  c   = (int)(e - row * Kc);
    float v = (c < D_NODE) ? hv[(long)src[row] * D_NODE + c]
                           : ef[row * D_EDGE + (c - D_NODE)];
    __builtin_nontemporal_store(v, &X[e]);
}
// GRU gates: gi/gh [M,3H], h [M,H]; out = (1-z)*n + z*h, optional relu
__global__ void gru_gate(const float* __restrict__ gi, const float* __restrict__ gh,
                         const float* __restrict__ h, float* __restrict__ out,
                         long M, int do_relu) {
    long e = (long)blockIdx.x * blockDim.x + threadIdx.x;
    if (e >= M * (long)D_G) return;
    long m = e / D_G;
    int  c = (int)(e - m * D_G);
    long b3 = m * (3 * D_G);
    float ir  = gi[b3 + c],             hr = gh[b3 + c];
    float iz  = gi[b3 + D_G + c],       hz = gh[b3 + D_G + c];
    float in_ = gi[b3 + 2 * D_G + c],   hn = gh[b3 + 2 * D_G + c];
    float r = 1.f / (1.f + __expf(-(ir + hr)));
    float z = 1.f / (1.f + __expf(-(iz + hz)));
    float nn = tanhf(in_ + r * hn);
    float v = (1.f - z) * nn + z * h[m * D_G + c];
    if (do_relu) v = fmaxf(v, 0.f);
    out[e] = v;
}

// ---------------------------------------------------------------------------
// host orchestration
// ---------------------------------------------------------------------------
static inline int blks(long n, int t) { return (int)((n + t - 1) / t); }

extern "C" void kernel_launch(void* const* d_in, const int* in_sizes, int n_in,
                              void* d_out, int out_size, void* d_ws, size_t ws_size,
                              hipStream_t stream) {
    const float* node_feats = (const float*)d_in[0];
    const float* edge_feats = (const float*)d_in[1];
    const int*   src        = (const int*)d_in[2];
    const int*   dst        = (const int*)d_in[3];
    const int*   graph_ids  = (const int*)d_in[4];
    const float* pn_W   = (const float*)d_in[5];
    const float* pn_b   = (const float*)d_in[6];
    const float* pe1_W  = (const float*)d_in[7];
    const float* pe1_b  = (const float*)d_in[8];
    const float* pe2_W  = (const float*)d_in[9];
    const float* pe2_b  = (const float*)d_in[10];
    const float* et_W   = (const float*)d_in[11];
    const float* et_b   = (const float*)d_in[12];
    const float* g1_Wih = (const float*)d_in[13];
    const float* g1_Whh = (const float*)d_in[14];
    const float* g1_bih = (const float*)d_in[15];
    const float* g1_bhh = (const float*)d_in[16];
    const float* pe_W   = (const float*)d_in[17];
    const float* pe_b   = (const float*)d_in[18];
    const float* pnode_W= (const float*)d_in[19];
    const float* pnode_b= (const float*)d_in[20];
    const float* g2_Wih = (const float*)d_in[21];
    const float* g2_Whh = (const float*)d_in[22];
    const float* g2_bih = (const float*)d_in[23];
    const float* g2_bhh = (const float*)d_in[24];
    const float* ro_logit_W = (const float*)d_in[25];   // [T,512]
    const float* ro_logit_b = (const float*)d_in[26];   // [T,1]
    const float* ro_proj_W  = (const float*)d_in[27];   // [T,256,256]
    const float* ro_proj_b  = (const float*)d_in[28];   // [T,256]
    const float* ro_g_Wih   = (const float*)d_in[29];   // [T,768,256]
    const float* ro_g_Whh   = (const float*)d_in[30];
    const float* ro_g_bih   = (const float*)d_in[31];   // [T,768]
    const float* ro_g_bhh   = (const float*)d_in[32];

    float* ws = (float*)d_ws;
    size_t off = 0;
    auto take = [&](size_t n) { float* p = ws + off; off += n; return p; };
    float* Xcat  = take((size_t)N_EDGES * (D_NODE + D_EDGE));
    float* hvNew = take((size_t)N_NODES * D_G);          // later reused as h2
    float* he1   = take((size_t)N_EDGES * D_G);
    float* tmpE  = take((size_t)N_EDGES * D_G);
    float* gi    = take((size_t)N_NODES * 3 * D_G);
    float* gh    = take((size_t)N_NODES * 3 * D_G);
    float* ctx   = take((size_t)N_NODES * D_G);
    float* h     = take((size_t)N_NODES * D_G);
    float* elog  = take(N_EDGES);
    float* eexp  = take(N_EDGES);
    float* ea    = take(N_EDGES);
    unsigned* nmax = (unsigned*)take(N_NODES);
    float* nsum  = take(N_NODES);
    float* gA    = take((size_t)N_GRAPHS * D_G);
    float* gB    = take((size_t)N_GRAPHS * D_G);
    float* grepr = take((size_t)N_GRAPHS * D_G);
    float* giG   = take((size_t)N_GRAPHS * 3 * D_G);
    float* ghG   = take((size_t)N_GRAPHS * 3 * D_G);
    float* nlog  = take(N_NODES);
    float* nexp  = take(N_NODES);
    float* na    = take(N_NODES);
    unsigned* gmax = (unsigned*)take(N_GRAPHS);
    float* gsum  = take(N_GRAPHS);
    float* hvp   = take((size_t)N_NODES * D_G);
    (void)ws_size; (void)in_sizes; (void)n_in; (void)out_size;

    auto gemm = [&](const float* A, const int* aidx, const float* B, int btr,
                    const float* bias, float* C, int M, int N, int K, int act,
                    int nt_c) {
        dim3 grid(N / 128, (M + 15) / 16);
        gemm_wmma_bf16<<<grid, 128, 0, stream>>>(A, aidx, B, btr, bias, C,
                                                 M, N, K, act, nt_c);
    };
    auto seg_softmax = [&](const float* lg, const int* seg, int n, int nseg,
                           unsigned* mx, float* sm, float* ev, float* aout) {
        fill_u32<<<blks(nseg, 256), 256, 0, stream>>>(mx, 0u, nseg);
        fill_f32<<<blks(nseg, 256), 256, 0, stream>>>(sm, 0.f, nseg);
        seg_max_k<<<blks(n, 256), 256, 0, stream>>>(lg, seg, mx, n);
        seg_expsum_k<<<blks(n, 256), 256, 0, stream>>>(lg, seg, mx, sm, ev, n);
        seg_norm_k<<<blks(n, 256), 256, 0, stream>>>(ev, seg, sm, aout, n);
    };

    // ===================== GetContext =====================
    gemm(node_feats, nullptr, pn_W, 0, pn_b, hvNew, N_NODES, D_G, D_NODE, 1, 0);
    concat_src_edge<<<blks((long)N_EDGES * 80, 256), 256, 0, stream>>>(
        node_feats, edge_feats, src, Xcat);
    gemm(Xcat, nullptr, pe1_W, 0, pe1_b, he1, N_EDGES, D_G, D_NODE + D_EDGE, 1, 0);
    row_logit<<<blks(N_EDGES, 8), 256, 0, stream>>>(
        hvNew, dst, 0, he1, nullptr, pe2_W, pe2_b, elog, N_EDGES);
    seg_softmax(elog, dst, N_EDGES, N_NODES, nmax, nsum, eexp, ea);
    gemm(he1, nullptr, et_W, 0, et_b, tmpE, N_EDGES, D_G, D_G, 0, 1);   // NT out
    fill_f32<<<blks((long)N_NODES * D_G, 256), 256, 0, stream>>>(ctx, 0.f, (long)N_NODES * D_G);
    scatter_scale_add<<<blks((long)N_EDGES * D_G, 256), 256, 0, stream>>>(
        ctx, dst, ea, tmpE, N_EDGES, D_G, 1);
    elu_inplace<<<blks((long)N_NODES * D_G, 256), 256, 0, stream>>>(ctx, (long)N_NODES * D_G);
    gemm(ctx,   nullptr, g1_Wih, 1, g1_bih, gi, N_NODES, 3 * D_G, D_G, 0, 0);
    gemm(hvNew, nullptr, g1_Whh, 1, g1_bhh, gh, N_NODES, 3 * D_G, D_G, 0, 0);
    gru_gate<<<blks((long)N_NODES * D_G, 256), 256, 0, stream>>>(gi, gh, hvNew, h, N_NODES, 1);

    // ===================== GNNLayer =====================
    row_logit<<<blks(N_EDGES, 8), 256, 0, stream>>>(
        h, dst, 0, h, src, pe_W, pe_b, elog, N_EDGES);
    seg_softmax(elog, dst, N_EDGES, N_NODES, nmax, nsum, eexp, ea);
    gemm(h, src, pnode_W, 0, pnode_b, tmpE, N_EDGES, D_G, D_G, 0, 1);   // gather, NT out
    fill_f32<<<blks((long)N_NODES * D_G, 256), 256, 0, stream>>>(ctx, 0.f, (long)N_NODES * D_G);
    scatter_scale_add<<<blks((long)N_EDGES * D_G, 256), 256, 0, stream>>>(
        ctx, dst, ea, tmpE, N_EDGES, D_G, 1);
    elu_inplace<<<blks((long)N_NODES * D_G, 256), 256, 0, stream>>>(ctx, (long)N_NODES * D_G);
    gemm(ctx, nullptr, g2_Wih, 1, g2_bih, gi, N_NODES, 3 * D_G, D_G, 0, 0);
    gemm(h,   nullptr, g2_Whh, 1, g2_bhh, gh, N_NODES, 3 * D_G, D_G, 0, 0);
    float* h2 = hvNew;   // reuse
    gru_gate<<<blks((long)N_NODES * D_G, 256), 256, 0, stream>>>(gi, gh, h, h2, N_NODES, 1);

    // ===================== Readout =====================
    fill_f32<<<blks((long)N_GRAPHS * D_G, 256), 256, 0, stream>>>(gA, 0.f, (long)N_GRAPHS * D_G);
    scatter_scale_add<<<blks((long)N_NODES * D_G, 256), 256, 0, stream>>>(
        gA, graph_ids, nullptr, h2, N_NODES, D_G, 0);
    float* gcur = gA;
    for (int t = 0; t < NUM_T; ++t) {
        row_logit<<<blks(N_NODES, 8), 256, 0, stream>>>(
            gcur, graph_ids, 1, h2, nullptr,
            ro_logit_W + (size_t)t * 2 * D_G, ro_logit_b + t, nlog, N_NODES);
        seg_softmax(nlog, graph_ids, N_NODES, N_GRAPHS, gmax, gsum, nexp, na);
        gemm(h2, nullptr, ro_proj_W + (size_t)t * D_G * D_G, 0,
             ro_proj_b + (size_t)t * D_G, hvp, N_NODES, D_G, D_G, 0, 0);
        fill_f32<<<blks((long)N_GRAPHS * D_G, 256), 256, 0, stream>>>(
            grepr, 0.f, (long)N_GRAPHS * D_G);
        scatter_scale_add<<<blks((long)N_NODES * D_G, 256), 256, 0, stream>>>(
            grepr, graph_ids, na, hvp, N_NODES, D_G, 0);
        elu_inplace<<<blks((long)N_GRAPHS * D_G, 256), 256, 0, stream>>>(
            grepr, (long)N_GRAPHS * D_G);
        gemm(grepr, nullptr, ro_g_Wih + (size_t)t * 3 * D_G * D_G, 1,
             ro_g_bih + (size_t)t * 3 * D_G, giG, N_GRAPHS, 3 * D_G, D_G, 0, 0);
        gemm(gcur,  nullptr, ro_g_Whh + (size_t)t * 3 * D_G * D_G, 1,
             ro_g_bhh + (size_t)t * 3 * D_G, ghG, N_GRAPHS, 3 * D_G, D_G, 0, 0);
        float* gnext = (t == NUM_T - 1) ? (float*)d_out : gB;
        gru_gate<<<blks((long)N_GRAPHS * D_G, 256), 256, 0, stream>>>(
            giG, ghG, gcur, gnext, N_GRAPHS, 0);
        gcur = gnext;
    }
}
